// SpeMamba_26319559590731
// MI455X (gfx1250) — matrix operand order repaired and checked by reading, not compiled, not run on hardware
//
#include <hip/hip_runtime.h>
#include <hip/hip_bf16.h>

typedef __attribute__((ext_vector_type(16))) __bf16 v16bf;
typedef __attribute__((ext_vector_type(8)))  float  v8f;

#define TID threadIdx.x

// ---------------- LDS layout (bytes) ----------------
// XF  : bf16[256][40]   A-operand for in_proj (rows = seq*8+t, k = d_model)
// WB  : bf16 weight B-operand buffer (reused per GEMM), stride 40 or 72
// ZB  : bf16[64][264]   z (gate), column-major (d-major) for packed scatter
// UR  : bf16[64][264]   raw u, column-major -> conv reads 8 timesteps per b128
// UC  : bf16[256][72]   conv+silu output (A operand), later gated-y
// DBL : f32 [36][264]   x_proj output (dt|B|C), column-major; alias OUT f32[32][264]
// YA  : f32 [256][64]   y_f + y_b accumulator
#define OFF_XF   0
#define OFF_WB   20480
#define OFF_ZB   30720
#define OFF_UR   64512
#define OFF_UC   98304
#define OFF_DBL  135168
#define OFF_YA   173184
#define OFF_ASM  238720
#define OFF_DTW  242816
#define OFF_DTB  243328
#define OFF_DVV  243584
#define OFF_CW   243840
#define OFF_CB   244352
#define OFF_RED  244608
#define SMEM_BYTES 244736

#define CSTRIDE 264   // column-major padded stride (elements)

__device__ __forceinline__ float fsig(float x) { return 1.f / (1.f + __expf(-x)); }

// A-matrix 16x32 bf16 fragment: lane half selects K {0-7,16-23} vs {8-15,24-31}
__device__ __forceinline__ v16bf ld_fragA(const __bf16* base, int stride, int row, int ko) {
  int lane = TID & 31;
  const __bf16* p = base + (size_t)(row + (lane & 15)) * stride + ko + ((lane & 16) ? 8 : 0);
  union { v16bf v; float4 f[2]; } u;
  u.f[0] = *(const float4*)(p);
  u.f[1] = *(const float4*)(p + 16);
  return u.v;
}
// B-matrix 32x16 bf16 fragment: lane half selects K {0-7,16-23} vs {8-15,24-31}, N = lane&15
__device__ __forceinline__ v16bf ld_fragB(const __bf16* base, int stride, int row, int ko) {
  int lane = TID & 31;
  const __bf16* p = base + (size_t)(row + (lane & 15)) * stride + ko + ((lane & 16) ? 8 : 0);
  union { v16bf v; float4 f[2]; } u;
  u.f[0] = *(const float4*)(p);
  u.f[1] = *(const float4*)(p + 16);
  return u.v;
}

// pack 8 f32 accumulator elements -> 8 bf16 -> single b128 LDS store
__device__ __forceinline__ void st_pack8_bf16(__bf16* dst, v8f acc) {
  union { __bf16 h[8]; float4 f4; } p;
  #pragma unroll
  for (int r = 0; r < 8; ++r) p.h[r] = (__bf16)acc[r];
  *(float4*)dst = p.f4;
}
// 8 f32 -> two b128 LDS stores
__device__ __forceinline__ void st_pack8_f32(float* dst, v8f acc) {
  *(float4*)dst       = make_float4(acc[0], acc[1], acc[2], acc[3]);
  *(float4*)(dst + 4) = make_float4(acc[4], acc[5], acc[6], acc[7]);
}

__global__ void spe_zero(float* ws) { ws[TID] = 0.f; }

__global__ void __launch_bounds__(256) spe_mamba_main(
    const float* __restrict__ x,
    const float* __restrict__ ipw,
    const float* __restrict__ cwf, const float* __restrict__ cbf,
    const float* __restrict__ cwb, const float* __restrict__ cbb,
    const float* __restrict__ xpf, const float* __restrict__ xpb,
    const float* __restrict__ dtwf, const float* __restrict__ dtbf,
    const float* __restrict__ dtwb, const float* __restrict__ dtbb,
    const float* __restrict__ alf, const float* __restrict__ alb,
    const float* __restrict__ Dfp, const float* __restrict__ Dbp,
    const float* __restrict__ opw,
    float* __restrict__ out, float* __restrict__ wsums) {
  extern __shared__ char smem[];
  __bf16* XF  = (__bf16*)(smem + OFF_XF);
  __bf16* WB  = (__bf16*)(smem + OFF_WB);
  __bf16* ZB  = (__bf16*)(smem + OFF_ZB);
  __bf16* UR  = (__bf16*)(smem + OFF_UR);
  __bf16* UC  = (__bf16*)(smem + OFF_UC);
  float*  DBL = (float*)(smem + OFF_DBL);
  float*  OUTB= (float*)(smem + OFF_DBL);   // alias (used after DBL is dead)
  float*  YA  = (float*)(smem + OFF_YA);
  float*  ASM = (float*)(smem + OFF_ASM);
  float*  DTW = (float*)(smem + OFF_DTW);
  float*  DTB = (float*)(smem + OFF_DTB);
  float*  DVV = (float*)(smem + OFF_DVV);
  float*  CW  = (float*)(smem + OFF_CW);
  float*  CB  = (float*)(smem + OFF_CB);
  float*  RED = (float*)(smem + OFF_RED);

  const int tid  = TID;
  const int wave = tid >> 5;
  const int lane = tid & 31;
  const int n0   = blockIdx.x * 32;           // base sequence
  const int b    = n0 >> 12;                  // / (64*64)
  const int rem  = n0 & 4095;
  const int hh   = rem >> 6;
  const int w0   = rem & 63;
  const float* xg = x + (size_t)b * 1048576 + hh * 64 + w0;

  __builtin_prefetch(ipw + tid * 16, 0, 0);   // global_prefetch_b8: warm weight lines

  // ---- Phase 0: load x tile (coalesced across w) + in_proj weights ----
  #pragma unroll 4
  for (int j = 0; j < 32; ++j) {
    int c = (tid >> 5) + 8 * j;               // channel 0..255
    float v = xg[(size_t)c * 4096 + (tid & 31)];
    XF[((tid & 31) * 8 + (c >> 5)) * 40 + (c & 31)] = (__bf16)v;
  }
  for (int j = tid; j < 4096; j += 256)
    WB[(j >> 5) * 40 + (j & 31)] = (__bf16)ipw[j];
  __syncthreads();

  // ---- Phase 1: in_proj GEMM  M=256 K=32 N=128 -> u | z (column-major scatter) ----
  {
    const int moff = (lane & 16) ? 8 : 0;
    const int n    = lane & 15;
    int m0 = wave * 32;
    v16bf a0 = ld_fragA(XF, 40, m0, 0);
    v16bf a1 = ld_fragA(XF, 40, m0 + 16, 0);
    #pragma unroll
    for (int nt = 0; nt < 8; ++nt) {
      v16bf bb = ld_fragB(WB, 40, nt * 16, 0);
      v8f acc0 = {}, acc1 = {};
      acc0 = __builtin_amdgcn_wmma_f32_16x16x32_bf16(false, a0, false, bb, (short)0, acc0, false, false);
      acc1 = __builtin_amdgcn_wmma_f32_16x16x32_bf16(false, a1, false, bb, (short)0, acc1, false, false);
      int col = nt * 16 + n;
      __bf16* dst = (nt < 4) ? (UR + col * CSTRIDE) : (ZB + (col - 64) * CSTRIDE);
      st_pack8_bf16(dst + m0 + moff,      acc0);
      st_pack8_bf16(dst + m0 + 16 + moff, acc1);
    }
  }
  __syncthreads();

  // ---- Branch loop: 0 = forward, 1 = backward (flipped time) ----
  for (int br = 0; br < 2; ++br) {
    const float* al  = br ? alb  : alf;
    const float* cw_ = br ? cwb  : cwf;
    const float* cb_ = br ? cbb  : cbf;
    const float* xp  = br ? xpb  : xpf;
    const float* dw  = br ? dtwb : dtwf;
    const float* db_ = br ? dtbb : dtbf;
    const float* Dv  = br ? Dbp  : Dfp;

    for (int j = tid; j < 1024; j += 256) ASM[j] = -__expf(al[j]);      // A = -exp(A_log)
    if (tid < 128) { DTW[tid] = dw[tid]; CW[tid] = cw_[tid]; }
    if (tid < 64)  { DTB[tid] = db_[tid]; CB[tid] = cb_[tid]; DVV[tid] = Dv[tid]; }
    for (int j = tid; j < 3072; j += 256) {                              // x_proj_w -> [n][k], pad n to 48
      int n = j >> 6, k = j & 63;
      WB[n * 72 + k] = (n < 34) ? (__bf16)xp[n * 64 + k] : (__bf16)0.f;
    }
    __syncthreads();

    // conv(2) causal + SiLU: one b128 load gives all 8 timesteps of a (seq,d) channel
    {
      int seq = tid >> 3;
      #pragma unroll
      for (int i = 0; i < 8; ++i) {
        int d = (tid & 7) * 8 + i;
        float w0c = CW[d * 2], w1c = CW[d * 2 + 1], bc = CB[d];
        union { float4 f4; __bf16 h[8]; } up;
        up.f4 = *(const float4*)(UR + d * CSTRIDE + seq * 8);
        float prev = 0.f;
        #pragma unroll
        for (int t = 0; t < 8; ++t) {
          float cur = (float)up.h[br ? (7 - t) : t];
          float v = w0c * prev + w1c * cur + bc;
          UC[(seq * 8 + t) * 72 + d] = (__bf16)(v * fsig(v));
          prev = cur;
        }
      }
    }
    __syncthreads();

    // x_proj GEMM  M=256 K=64 N=48(34 used) -> DBL (f32, column-major)
    {
      const int moff = (lane & 16) ? 8 : 0;
      const int n    = lane & 15;
      v16bf wb0[3], wb1[3];
      #pragma unroll
      for (int nt = 0; nt < 3; ++nt) {
        wb0[nt] = ld_fragB(WB, 72, nt * 16, 0);
        wb1[nt] = ld_fragB(WB, 72, nt * 16, 32);
      }
      #pragma unroll
      for (int mt = 0; mt < 2; ++mt) {
        int m0 = (wave * 2 + mt) * 16;
        v16bf a0 = ld_fragA(UC, 72, m0, 0);
        v16bf a1 = ld_fragA(UC, 72, m0, 32);
        #pragma unroll
        for (int nt = 0; nt < 3; ++nt) {
          v8f acc = {};
          acc = __builtin_amdgcn_wmma_f32_16x16x32_bf16(false, a0, false, wb0[nt], (short)0, acc, false, false);
          acc = __builtin_amdgcn_wmma_f32_16x16x32_bf16(false, a1, false, wb1[nt], (short)0, acc, false, false);
          int col = nt * 16 + n;
          if (col < 36) st_pack8_f32(DBL + col * CSTRIDE + m0 + moff, acc);
        }
      }
    }
    __syncthreads();

    // selective scan: each thread owns (seq, 8 d-channels), 16-state recurrence in regs
    {
      int seq = tid >> 3;
      int d0 = (tid & 7) * 8;
      for (int dd = 0; dd < 8; ++dd) {
        int d = d0 + dd;
        float Ar[16];
        #pragma unroll
        for (int s = 0; s < 16; ++s) Ar[s] = ASM[d * 16 + s];
        float w0d = DTW[d * 2], w1d = DTW[d * 2 + 1], bdt = DTB[d], Dd = DVV[d];
        float hst[16];
        #pragma unroll
        for (int s = 0; s < 16; ++s) hst[s] = 0.f;
        for (int t = 0; t < 8; ++t) {
          int row = seq * 8 + t;
          float xdt = DBL[row] * w0d + DBL[CSTRIDE + row] * w1d + bdt;
          float delta = (xdt > 20.f) ? xdt : __logf(1.f + __expf(xdt));   // softplus
          float u = (float)UC[row * 72 + d];
          float du = delta * u;
          float y = Dd * u;
          #pragma unroll
          for (int s = 0; s < 16; ++s) {
            float Bs = DBL[(2 + s) * CSTRIDE + row];
            float Cs = DBL[(18 + s) * CSTRIDE + row];
            hst[s] = __expf(delta * Ar[s]) * hst[s] + du * Bs;
            y += hst[s] * Cs;
          }
          int to = br ? (7 - t) : t;                                      // flip back for backward
          float* yp = &YA[(seq * 8 + to) * 64 + d];
          if (br) *yp += y; else *yp = y;
        }
      }
    }
    __syncthreads();
  }

  // ---- Phase 2: y = (y_f + y_b) * silu(z) -> UC (bf16, A operand); load out_proj_w ----
  for (int j = tid; j < 16384; j += 256) {
    int row = j >> 6, d = j & 63;
    float z = (float)ZB[d * CSTRIDE + row];
    UC[row * 72 + d] = (__bf16)(YA[row * 64 + d] * z * fsig(z));
  }
  for (int j = tid; j < 2048; j += 256) {
    int n = j >> 6, k = j & 63;
    WB[n * 72 + k] = (__bf16)opw[j];
  }
  __syncthreads();

  // ---- Phase 3: out_proj GEMM  M=256 K=64 N=32 -> OUTB (f32, column-major) ----
  {
    const int moff = (lane & 16) ? 8 : 0;
    const int n    = lane & 15;
    v16bf wb0[2], wb1[2];
    #pragma unroll
    for (int nt = 0; nt < 2; ++nt) {
      wb0[nt] = ld_fragB(WB, 72, nt * 16, 0);
      wb1[nt] = ld_fragB(WB, 72, nt * 16, 32);
    }
    #pragma unroll
    for (int mt = 0; mt < 2; ++mt) {
      int m0 = (wave * 2 + mt) * 16;
      v16bf a0 = ld_fragA(UC, 72, m0, 0);
      v16bf a1 = ld_fragA(UC, 72, m0, 32);
      #pragma unroll
      for (int nt = 0; nt < 2; ++nt) {
        v8f acc = {};
        acc = __builtin_amdgcn_wmma_f32_16x16x32_bf16(false, a0, false, wb0[nt], (short)0, acc, false, false);
        acc = __builtin_amdgcn_wmma_f32_16x16x32_bf16(false, a1, false, wb1[nt], (short)0, acc, false, false);
        st_pack8_f32(OUTB + (nt * 16 + n) * CSTRIDE + m0 + moff, acc);
      }
    }
  }
  __syncthreads();

  // ---- Phase 4: coalesced global store of pre-GN output + group sum/sumsq reduction ----
  if (tid < 8) RED[tid] = 0.f;
  __syncthreads();
  {
    int s = tid & 31;
    size_t gbase = (size_t)b * 1048576 + hh * 64 + w0 + s;
    #pragma unroll
    for (int g = 0; g < 4; ++g) {
      float sv = 0.f, qv = 0.f;
      #pragma unroll
      for (int jj = 0; jj < 8; ++jj) {
        int c = (tid >> 5) + 64 * g + 8 * jj;   // channel; group(c) == g by construction
        float val = OUTB[(c & 31) * CSTRIDE + s * 8 + (c >> 5)];
        out[gbase + (size_t)c * 4096] = val;
        sv += val; qv += val * val;
      }
      atomicAdd(&RED[g], sv);
      atomicAdd(&RED[4 + g], qv);
    }
  }
  __syncthreads();
  if (tid < 4)       atomicAdd(&wsums[b * 4 + tid], RED[tid]);
  else if (tid < 8)  atomicAdd(&wsums[32 + b * 4 + (tid - 4)], RED[tid]);
}

// ---- GroupNorm finalize + residual + SiLU (in place on d_out) ----
__global__ void __launch_bounds__(256) spe_gn(const float* __restrict__ x,
                                              const float* __restrict__ gnw,
                                              const float* __restrict__ gnb,
                                              float* __restrict__ out,
                                              const float* __restrict__ ws) {
  int idx = blockIdx.x * 256 + TID;
  int b = idx >> 20;
  int c = (idx >> 12) & 255;
  int g = (b << 2) | (c >> 6);
  const float inv = 1.f / 262144.f;            // 64 channels * 64 * 64 spatial
  float mu  = ws[g] * inv;
  float var = ws[32 + g] * inv - mu * mu;
  float v = out[idx];
  float gn = (v - mu) * rsqrtf(var + 1e-5f) * gnw[c] + gnb[c];
  out[idx] = x[idx] + gn * fsig(gn);
}

extern "C" void kernel_launch(void* const* d_in, const int* in_sizes, int n_in,
                              void* d_out, int out_size, void* d_ws, size_t ws_size,
                              hipStream_t stream) {
  const float* x    = (const float*)d_in[0];
  const float* ipw  = (const float*)d_in[1];
  const float* cwf  = (const float*)d_in[2];
  const float* cbf  = (const float*)d_in[3];
  const float* cwb  = (const float*)d_in[4];
  const float* cbb  = (const float*)d_in[5];
  const float* xpf  = (const float*)d_in[6];
  const float* xpb  = (const float*)d_in[7];
  const float* dtwf = (const float*)d_in[8];
  const float* dtbf = (const float*)d_in[9];
  const float* dtwb = (const float*)d_in[10];
  const float* dtbb = (const float*)d_in[11];
  const float* alf  = (const float*)d_in[12];
  const float* alb  = (const float*)d_in[13];
  const float* Df   = (const float*)d_in[14];
  const float* Db   = (const float*)d_in[15];
  const float* opw  = (const float*)d_in[16];
  const float* gnw  = (const float*)d_in[17];
  const float* gnb  = (const float*)d_in[18];
  float* out = (float*)d_out;
  float* ws  = (float*)d_ws;

  (void)hipFuncSetAttribute((const void*)spe_mamba_main,
                            hipFuncAttributeMaxDynamicSharedMemorySize, SMEM_BYTES);

  spe_zero<<<1, 64, 0, stream>>>(ws);
  spe_mamba_main<<<1024, 256, SMEM_BYTES, stream>>>(
      x, ipw, cwf, cbf, cwb, cbb, xpf, xpb,
      dtwf, dtbf, dtwb, dtbb, alf, alb, Df, Db, opw, out, ws);
  spe_gn<<<32768, 256, 0, stream>>>(x, gnw, gnb, out, ws);
}
